// EdgeDegreeEmbedding_2164663517596
// MI455X (gfx1250) — matrix operand order repaired and checked by reading, not compiled, not run on hardware
//
#include <hip/hip_runtime.h>
#include <hip/hip_bf16.h>
#include <stdint.h>

typedef __attribute__((ext_vector_type(16))) _Float16 v16h;
typedef __attribute__((ext_vector_type(8)))  float    v8f;

#define D_RBF   64
#define HDIM    128
#define OUTD    256
#define NUMI    16      // NUM_FULL_COEFFS
#define NUMJ    4       // NUM_M0
#define CDIM    64
#define WIG_L   14      // last dim of wigner_inv
#define STR     136     // padded LDS row stride in halfs (272B -> 4-bank rotation, conflict free)
#define WAVES   4
#define TPB     (WAVES * 32)
#define TILE_E  16

union FragU { v16h v; uint32_t u[8]; };

// A fragment (16x32 f16) from LDS row-major tile [16][STR].
// Layout used here (matches frag_b): lane m%16 = row; lanes<16 hold K=0..15, lanes>=16 hold K=16..31 (contiguous)
__device__ __forceinline__ v16h frag_a(const _Float16* hb, int lane, int k0) {
    const int m  = lane & 15;
    const int kb = (lane >> 4) * 16;           // 0 or 16
    const _Float16* p = hb + m * STR + k0 + kb;
    FragU f;
#pragma unroll
    for (int v = 0; v < 4; ++v) f.u[v]     = *(const uint32_t*)(p + 2 * v);
#pragma unroll
    for (int v = 0; v < 4; ++v) f.u[4 + v] = *(const uint32_t*)(p + 8 + 2 * v);
    return f.v;
}

// B fragment (32x16 f16) from N-major (transposed) weights, row stride = kstride.
// Layout used here (matches frag_a): lane = column n; lanes<16 hold K=0..15, lanes>=16 hold K=16..31 (contiguous).
__device__ __forceinline__ v16h frag_b(const _Float16* wT, int n0, int kstride, int lane, int k0) {
    const int n  = n0 + (lane & 15);
    const int kb = (lane >> 4) * 16;           // 0 or 16
    const _Float16* p = wT + (size_t)n * kstride + k0 + kb;
    FragU f;
#pragma unroll
    for (int v = 0; v < 8; ++v) f.u[v] = *(const uint32_t*)(p + 2 * v);
    return f.v;
}

// bias-add -> LayerNorm -> silu on C-layout accumulators (ntiles*16 = feature dim),
// result written as f16 row-major into the wave's LDS tile.
__device__ __forceinline__ void bias_ln_silu_store(v8f* acc, int ntiles,
                                                   const float* __restrict__ bia,
                                                   const float* __restrict__ gam,
                                                   const float* __restrict__ bet,
                                                   _Float16* hb, int lane) {
    const int col   = lane & 15;
    const int rbase = (lane >> 4) * 8;
    const float invH = 1.0f / (float)(16 * ntiles);
    float bv[8], gv[8], ev[8];
#pragma unroll
    for (int t = 0; t < 8; ++t) {
        if (t < ntiles) {
            int n = 16 * t + col;
            bv[t] = bia[n]; gv[t] = gam[n]; ev[t] = bet[n];
        }
    }
#pragma unroll
    for (int r = 0; r < 8; ++r) {
        float s = 0.f, s2 = 0.f;
#pragma unroll
        for (int t = 0; t < 8; ++t) {
            if (t < ntiles) {
                float x = acc[t][r] + bv[t];
                s += x; s2 += x * x;
            }
        }
#pragma unroll
        for (int off = 8; off >= 1; off >>= 1) {   // reduce within 16-lane half
            s  += __shfl_xor(s,  off, 32);
            s2 += __shfl_xor(s2, off, 32);
        }
        const float mu   = s * invH;
        const float var  = s2 * invH - mu * mu;
        const float rstd = rsqrtf(var + 1e-5f);
        const int   m    = rbase + r;
#pragma unroll
        for (int t = 0; t < 8; ++t) {
            if (t < ntiles) {
                int n = 16 * t + col;
                float x = acc[t][r] + bv[t];
                float y = (x - mu) * rstd * gv[t] + ev[t];
                float h = y / (1.0f + __expf(-y));        // silu
                hb[m * STR + n] = (_Float16)h;
            }
        }
    }
}

__global__ __launch_bounds__(TPB)
void edge_mlp_scatter(const float* __restrict__ x_edge,
                      const float* __restrict__ wig,
                      const _Float16* __restrict__ ws,   // W1T[128][64] | W2T[128][128] | W3T[256][128]
                      const float* __restrict__ b1, const float* __restrict__ g1, const float* __restrict__ be1,
                      const float* __restrict__ b2, const float* __restrict__ g2, const float* __restrict__ be2,
                      const float* __restrict__ b3,
                      const int* __restrict__ edge_dst,
                      float* __restrict__ out,
                      int E) {
    __shared__ __align__(16) _Float16 s_h[WAVES][TILE_E][STR];
    __shared__ __align__(16) float    s_w[WAVES][TILE_E][NUMI][NUMJ];
    __shared__ int                    s_dst[WAVES][TILE_E];

    const int lane = threadIdx.x & 31;
    const int wv   = threadIdx.x >> 5;
    const int e0   = (blockIdx.x * WAVES + wv) * TILE_E;   // wave-uniform
    if (e0 >= E) return;                                   // whole wave exits together

    _Float16* hb = &s_h[wv][0][0];

    // ---- prefetch this wave's wigner rows (16 edges x 896B = 112 cachelines) ----
    // Warms L2/L0 while GEMM1+GEMM2 run; actual gather is staged after GEMM2.
    {
        const size_t wig_bytes_per_edge = (size_t)NUMI * WIG_L * sizeof(float); // 896
        const char* wbase = (const char*)(wig + (size_t)e0 * NUMI * WIG_L);
        const size_t tile_bytes = (size_t)TILE_E * wig_bytes_per_edge;          // 14336
#pragma unroll
        for (int it = 0; it < 4; ++it) {
            size_t off = ((size_t)(it * 32 + lane)) * 128;                      // cacheline stride
            if (off < tile_bytes && (size_t)e0 * wig_bytes_per_edge + off <
                                    (size_t)E * wig_bytes_per_edge)
                __builtin_prefetch(wbase + off, 0, 0);                          // global_prefetch_b8
        }
    }

    // ---- stage x tile (16 edges x 64 rbf) into LDS as f16 ----
#pragma unroll
    for (int it = 0; it < (TILE_E * D_RBF) / 32; ++it) {
        int idx = it * 32 + lane;
        int e = idx >> 6, c = idx & 63;
        int eg = e0 + e; if (eg >= E) eg = E - 1;
        hb[e * STR + c] = (_Float16)x_edge[(size_t)eg * D_RBF + c];
    }

    v8f acc[16];

    // ================= GEMM1: (16x64) @ W1(64x128) =================
    const _Float16* W1T = ws;                       // [128][64]
#pragma unroll
    for (int t = 0; t < 8; ++t)
#pragma unroll
        for (int q = 0; q < 8; ++q) acc[t][q] = 0.f;
#pragma unroll
    for (int k0 = 0; k0 < 64; k0 += 32) {
        v16h a = frag_a(hb, lane, k0);
#pragma unroll
        for (int t = 0; t < 8; ++t) {
            v16h b = frag_b(W1T, 16 * t, 64, lane, k0);
            acc[t] = __builtin_amdgcn_wmma_f32_16x16x32_f16(false, a, false, b,
                                                            (short)0, acc[t], false, false);
        }
    }
    bias_ln_silu_store(acc, 8, b1, g1, be1, hb, lane);

    // ================= GEMM2: (16x128) @ W2(128x128) =================
    const _Float16* W2T = ws + 128 * 64;            // [128][128]
#pragma unroll
    for (int t = 0; t < 8; ++t)
#pragma unroll
        for (int q = 0; q < 8; ++q) acc[t][q] = 0.f;
#pragma unroll
    for (int k0 = 0; k0 < 128; k0 += 32) {
        v16h a = frag_a(hb, lane, k0);
#pragma unroll
        for (int t = 0; t < 8; ++t) {
            v16h b = frag_b(W2T, 16 * t, 128, lane, k0);
            acc[t] = __builtin_amdgcn_wmma_f32_16x16x32_f16(false, a, false, b,
                                                            (short)0, acc[t], false, false);
        }
    }
    bias_ln_silu_store(acc, 8, b2, g2, be2, hb, lane);

    // ---- stage selected wigner cols {0,2,6,11} : [16e][16i][4j] (cache-warm now);
    //      residual latency overlaps with GEMM3 issue below ----
#pragma unroll
    for (int it = 0; it < (TILE_E * NUMI * NUMJ) / 32; ++it) {
        int idx = it * 32 + lane;
        int e = idx >> 6;
        int i = (idx >> 2) & 15;
        int j = idx & 3;
        int mj = (j == 0) ? 0 : (j == 1) ? 2 : (j == 2) ? 6 : 11;
        int eg = e0 + e; if (eg >= E) eg = E - 1;
        s_w[wv][e][i][j] = wig[((size_t)eg * NUMI + i) * WIG_L + mj];
    }
    if (lane < TILE_E) {
        int eg = e0 + lane; if (eg >= E) eg = E - 1;
        s_dst[wv][lane] = edge_dst[eg];
    }

    // ================= GEMM3: (16x128) @ W3(128x256) =================
    const _Float16* W3T = ws + 128 * 64 + 128 * 128; // [256][128]
#pragma unroll
    for (int t = 0; t < 16; ++t)
#pragma unroll
        for (int q = 0; q < 8; ++q) acc[t][q] = 0.f;
#pragma unroll
    for (int k0 = 0; k0 < 128; k0 += 32) {
        v16h a = frag_a(hb, lane, k0);
#pragma unroll
        for (int t = 0; t < 16; ++t) {
            v16h b = frag_b(W3T, 16 * t, 128, lane, k0);
            acc[t] = __builtin_amdgcn_wmma_f32_16x16x32_f16(false, a, false, b,
                                                            (short)0, acc[t], false, false);
        }
    }

    // ===== einsum w(16x4) @ m0(4x64) straight from accumulators + scatter =====
    // h3 col n = j*64 + c ; c-tile t holds c = 16t+col ; j-th value lives in tile 4j+t.
    const int col   = lane & 15;
    const int rbase = (lane >> 4) * 8;
#pragma unroll
    for (int t = 0; t < 4; ++t) {
        const int c = 16 * t + col;
        const float bj0 = b3[c], bj1 = b3[64 + c], bj2 = b3[128 + c], bj3 = b3[192 + c];
#pragma unroll
        for (int r = 0; r < 8; ++r) {
            const int m  = rbase + r;
            const int eg = e0 + m;
            if (eg < E) {
                const float v0 = acc[t][r]      + bj0;
                const float v1 = acc[t + 4][r]  + bj1;
                const float v2 = acc[t + 8][r]  + bj2;
                const float v3 = acc[t + 12][r] + bj3;
                float* op = out + (size_t)s_dst[wv][m] * (NUMI * CDIM) + c;
#pragma unroll 4
                for (int i = 0; i < NUMI; ++i) {
                    float4 w = *(const float4*)&s_w[wv][m][i][0];   // LDS broadcast
                    float val = v0 * w.x + v1 * w.y + v2 * w.z + v3 * w.w;
                    atomicAdd(op + i * CDIM, val * 0.2f);           // /RESCALE_FACTOR
                }
            }
        }
    }
}

__global__ void zero_out_kernel(float* __restrict__ out, int n) {
    int i = blockIdx.x * blockDim.x + threadIdx.x;
    if (i < n) out[i] = 0.f;
}

// Convert W1/W2/W3 to f16, transposed to N-major for contiguous-K B-fragment loads.
__global__ void prep_weights_kernel(const float* __restrict__ W1,
                                    const float* __restrict__ W2,
                                    const float* __restrict__ W3,
                                    _Float16* __restrict__ ws) {
    int i = blockIdx.x * blockDim.x + threadIdx.x;
    if (i < 128 * 64) {                       // W1T[128][64]
        int n = i / 64, k = i % 64;
        ws[i] = (_Float16)W1[k * 128 + n];
    } else if (i < 128 * 64 + 128 * 128) {    // W2T[128][128]
        int i2 = i - 128 * 64;
        int n = i2 / 128, k = i2 % 128;
        ws[i] = (_Float16)W2[k * 128 + n];
    } else if (i < 128 * 64 + 128 * 128 + 256 * 128) {  // W3T[256][128]
        int i3 = i - 128 * 64 - 128 * 128;
        int n = i3 / 128, k = i3 % 128;
        ws[i] = (_Float16)W3[k * 256 + n];
    }
}

extern "C" void kernel_launch(void* const* d_in, const int* in_sizes, int n_in,
                              void* d_out, int out_size, void* d_ws, size_t ws_size,
                              hipStream_t stream) {
    const float* x_edge  = (const float*)d_in[0];
    const float* wig     = (const float*)d_in[1];
    const float* W1      = (const float*)d_in[2];
    const float* b1      = (const float*)d_in[3];
    const float* g1      = (const float*)d_in[4];
    const float* be1     = (const float*)d_in[5];
    const float* W2      = (const float*)d_in[6];
    const float* b2      = (const float*)d_in[7];
    const float* g2      = (const float*)d_in[8];
    const float* be2     = (const float*)d_in[9];
    const float* W3      = (const float*)d_in[10];
    const float* b3      = (const float*)d_in[11];
    const int*   edge_dst = (const int*)d_in[12];
    float*    out = (float*)d_out;
    _Float16* ws  = (_Float16*)d_ws;

    const int E = in_sizes[0] / D_RBF;

    zero_out_kernel<<<(out_size + 255) / 256, 256, 0, stream>>>(out, out_size);

    const int wtot = 128 * 64 + 128 * 128 + 256 * 128;
    prep_weights_kernel<<<(wtot + 255) / 256, 256, 0, stream>>>(W1, W2, W3, ws);

    const int blocks = (E + WAVES * TILE_E - 1) / (WAVES * TILE_E);
    edge_mlp_scatter<<<blocks, TPB, 0, stream>>>(x_edge, wig, ws,
                                                 b1, g1, be1, b2, g2, be2, b3,
                                                 edge_dst, out, E);
}
